// ESN_49598282334265
// MI455X (gfx1250) — matrix operand order repaired and checked by reading, not compiled, not run on hardware
//
#include <hip/hip_runtime.h>
#include <hip/hip_bf16.h>

#define SEQ_LEN   2048
#define STATE_DIM 128
#define RES_DIM   4096

typedef __attribute__((ext_vector_type(16))) __bf16 v16bf;
typedef __attribute__((ext_vector_type(8)))  float  v8f;

__device__ inline v16bf cvt16(float4 a, float4 b, float4 c, float4 d) {
    v16bf r;
    r[0]  = (__bf16)a.x; r[1]  = (__bf16)a.y; r[2]  = (__bf16)a.z; r[3]  = (__bf16)a.w;
    r[4]  = (__bf16)b.x; r[5]  = (__bf16)b.y; r[6]  = (__bf16)b.z; r[7]  = (__bf16)b.w;
    r[8]  = (__bf16)c.x; r[9]  = (__bf16)c.y; r[10] = (__bf16)c.z; r[11] = (__bf16)c.w;
    r[12] = (__bf16)d.x; r[13] = (__bf16)d.y; r[14] = (__bf16)d.z; r[15] = (__bf16)d.w;
    return r;
}

// ---------------------------------------------------------------------------
// Kernel 1: W_res fp32 -> bf16 in WMMA A-fragment order (contiguous k order:
// k = 16*(L/16) + s). Scan then streams W with one contiguous 32B load per lane; each
// wave-level load covers a contiguous 1KB region of L2.
// ---------------------------------------------------------------------------
__global__ __launch_bounds__(256) void wres_to_frag(const float* __restrict__ W,
                                                    __bf16* __restrict__ Wf) {
    size_t g = (size_t)blockIdx.x * 256 + threadIdx.x;   // fragment-lane id
    const int L  = (int)(g & 31);
    const int kt = (int)((g >> 5) & 127);
    const int rt = (int)(g >> 12);
    const int row = rt * 16 + (L & 15);
    const int kb  = kt * 32 + (L >> 4) * 16;
    const float* src = W + (size_t)row * RES_DIM;
    const float4* p0 = (const float4*)(src + kb);
    const float4* p1 = (const float4*)(src + kb + 8);
    *(v16bf*)(Wf + g * 16) = cvt16(p0[0], p0[1], p1[0], p1[1]);
}

// ---------------------------------------------------------------------------
// Kernel 2: proj = in_seq @ W_in^T + b  via v_wmma_f32_16x16x32_bf16.
// One wave per 16x16 tile; K=128 -> 4 WMMA steps, bias fused at store.
// ---------------------------------------------------------------------------
__global__ __launch_bounds__(256) void proj_gemm(const float* __restrict__ X,
                                                 const float* __restrict__ Win,
                                                 const float* __restrict__ bias,
                                                 float* __restrict__ proj) {
    const int lane = threadIdx.x & 31;
    const int wave = threadIdx.x >> 5;
    const int tile = blockIdx.x * 8 + wave;
    const int tn   = tile & (RES_DIM / 16 - 1);
    const int tm   = tile >> 8;
    const int hi   = lane >> 4;
    const int lm   = lane & 15;

    const float* xrow = X   + (size_t)(tm * 16 + lm) * STATE_DIM;
    const float* wrow = Win + (size_t)(tn * 16 + lm) * STATE_DIM;

    v8f acc = {};
#pragma unroll
    for (int kb = 0; kb < STATE_DIM; kb += 32) {
        const float4* pa0 = (const float4*)(xrow + kb + hi * 16);
        const float4* pa1 = (const float4*)(xrow + kb + hi * 16 + 8);
        v16bf af = cvt16(pa0[0], pa0[1], pa1[0], pa1[1]);
        const float4* pb = (const float4*)(wrow + kb + hi * 16);
        v16bf bf = cvt16(pb[0], pb[1], pb[2], pb[3]);
        acc = __builtin_amdgcn_wmma_f32_16x16x32_bf16(false, af, false, bf,
                                                      (short)0, acc, false, false);
    }

    const int col = tn * 16 + lm;
    const float bc = bias[col];
#pragma unroll
    for (int v = 0; v < 8; v++) {
        const int row = tm * 16 + hi * 8 + v;
        proj[(size_t)row * RES_DIM + col] = acc[v] + bc;
    }
}

// ---------------------------------------------------------------------------
// Kernel 3: one recurrence step: out = tanh(W @ r + proj_t).
// 256 blocks x 8 waves; block owns 16 rows, waves split K 8 ways. K-loop is
// HAND-UNROLLED x4 so 8 b128 loads are in flight per 4 chained WMMAs.
// ---------------------------------------------------------------------------
__global__ __launch_bounds__(256) void esn_step(const __bf16* __restrict__ Wf,
                                                const float*  __restrict__ r_prev_f32,
                                                const __bf16* __restrict__ r_prev_bf,
                                                const float*  __restrict__ projt,
                                                float* __restrict__ out,
                                                __bf16* __restrict__ r_next_bf) {
    __shared__ __align__(32) __bf16 r_lds[RES_DIM];      // 8 KB
    __shared__ float partial[8 * 16];

    const int tid = threadIdx.x;
    if (r_prev_bf) {                                     // steady state: raw copy
        ((v16bf*)r_lds)[tid] = ((const v16bf*)r_prev_bf)[tid];
    } else {                                             // t==0: convert res_state
        for (int i = tid; i < RES_DIM; i += 256)
            r_lds[i] = (__bf16)r_prev_f32[i];
    }
    __syncthreads();

    const int lane    = tid & 31;
    const int wave    = tid >> 5;
    const int hi      = lane >> 4;
    const int lm      = lane & 15;
    const int rowBase = blockIdx.x * 16;

    // fragment stream for this block's row tile; 32B per lane per K-tile
    const v16bf* wfrag = (const v16bf*)(Wf + ((size_t)blockIdx.x * 128 * 32 + lane) * 16);
    const v16bf* rvec  = (const v16bf*)&r_lds[hi * 16];  // + kt*2 per K-tile

    v8f acc = {};
    const int ktBase = wave * 16;                        // 16 of 128 K-tiles
    for (int kt = ktBase; kt < ktBase + 16; kt += 4) {
        v16bf a0 = wfrag[(size_t)(kt + 0) * 32];
        v16bf a1 = wfrag[(size_t)(kt + 1) * 32];
        v16bf a2 = wfrag[(size_t)(kt + 2) * 32];
        v16bf a3 = wfrag[(size_t)(kt + 3) * 32];
        v16bf b0 = rvec[(kt + 0) * 2];
        v16bf b1 = rvec[(kt + 1) * 2];
        v16bf b2 = rvec[(kt + 2) * 2];
        v16bf b3 = rvec[(kt + 3) * 2];
        acc = __builtin_amdgcn_wmma_f32_16x16x32_bf16(false, a0, false, b0, (short)0, acc, false, false);
        acc = __builtin_amdgcn_wmma_f32_16x16x32_bf16(false, a1, false, b1, (short)0, acc, false, false);
        acc = __builtin_amdgcn_wmma_f32_16x16x32_bf16(false, a2, false, b2, (short)0, acc, false, false);
        acc = __builtin_amdgcn_wmma_f32_16x16x32_bf16(false, a3, false, b3, (short)0, acc, false, false);
    }

    if (lm == 0) {                                       // harvest column 0
#pragma unroll
        for (int v = 0; v < 8; v++)
            partial[wave * 16 + hi * 8 + v] = acc[v];
    }
    __syncthreads();

    if (tid < 16) {
        float s = 0.f;
#pragma unroll
        for (int w = 0; w < 8; w++) s += partial[w * 16 + tid];
        const float val = tanhf(s + projt[rowBase + tid]);
        out[rowBase + tid]       = val;
        r_next_bf[rowBase + tid] = (__bf16)val;
    }
}

// ---------------------------------------------------------------------------
extern "C" void kernel_launch(void* const* d_in, const int* in_sizes, int n_in,
                              void* d_out, int out_size, void* d_ws, size_t ws_size,
                              hipStream_t stream) {
    const float* in_seq    = (const float*)d_in[0];   // 2048*128
    const float* res_state = (const float*)d_in[1];   // 4096 (zeros)
    const float* W_in      = (const float*)d_in[2];   // 4096*128
    const float* W_res     = (const float*)d_in[3];   // 4096*4096
    const float* b         = (const float*)d_in[4];   // 4096
    float* out = (float*)d_out;                       // 2048*4096

    const size_t wbytes = (size_t)RES_DIM * RES_DIM * sizeof(__bf16);   // 32 MB
    const size_t pbytes = (size_t)SEQ_LEN * RES_DIM * sizeof(float);    // 32 MB
    __bf16* Wf   = (__bf16*)d_ws;
    float*  proj = (float*)((char*)d_ws + wbytes);
    __bf16* r_bf = (__bf16*)((char*)d_ws + wbytes + pbytes);            // 8 KB

    // 1) W_res -> bf16 in WMMA-fragment order
    wres_to_frag<<<(RES_DIM / 16) * (RES_DIM / 32) * 32 / 256, 256, 0, stream>>>(W_res, Wf);

    // 2) proj = in_seq @ W_in^T + b
    proj_gemm<<<(SEQ_LEN / 16) * (RES_DIM / 16) / 8, 256, 0, stream>>>(in_seq, W_in, b, proj);

    // 3) sequential scan; bf16 state handed forward through workspace
    for (int t = 0; t < SEQ_LEN; t++) {
        esn_step<<<RES_DIM / 16, 256, 0, stream>>>(
            Wf, res_state, (t == 0) ? nullptr : r_bf,
            proj + (size_t)t * RES_DIM,
            out + (size_t)t * RES_DIM,
            r_bf);
    }
}
